// CausalSelfAttention_11330123726870
// MI455X (gfx1250) — matrix operand order repaired and checked by reading, not compiled, not run on hardware
//
#include <hip/hip_runtime.h>
#include <hip/hip_bf16.h>

typedef __attribute__((ext_vector_type(16))) __bf16 v16bf;
typedef __attribute__((ext_vector_type(8)))  __bf16 v8bf;
typedef __attribute__((ext_vector_type(4)))  __bf16 v4bf;
typedef __attribute__((ext_vector_type(8)))  float  v8f;

static constexpr int Bc = 2, Sc = 2048, Dc = 2048, Hc = 16, HDc = 128;

#define WMMA_BF16(a, b, c) \
  __builtin_amdgcn_wmma_f32_16x16x32_bf16(false, (a), false, (b), (short)0, (c), false, false)

__device__ __forceinline__ v8f v8f_zero() {
  v8f z;
#pragma unroll
  for (int i = 0; i < 8; ++i) z[i] = 0.0f;
  return z;
}

// A fragment (16x32 bf16, MxK), source row-major with leading dim ld.
// Layout: lanes 0-15 -> M=lane, K=0..15; lanes 16-31 -> K=16..31 (contiguous, matches B).
__device__ __forceinline__ v16bf load_a_frag(const __bf16* __restrict__ A,
                                             int row0, int k0, int ld, int lane) {
  int m    = row0 + (lane & 15);
  int koff = k0 + ((lane >> 4) << 4);
  const __bf16* p = A + (size_t)m * ld + koff;
  v8bf lo = *(const v8bf*)p;          // K = koff .. koff+7
  v8bf hi = *(const v8bf*)(p + 8);    // K = koff+8 .. koff+15
  v16bf r;
#pragma unroll
  for (int i = 0; i < 8; ++i) { r[i] = lo[i]; r[i + 8] = hi[i]; }
  return r;
}

// B fragment (32x16 bf16, KxN) where B[k][n] = W[n][k] (W row-major N x K, leading dim ld).
// Layout used here: lane n holds column n; lanes 0-15 K=0..15, lanes 16-31 K=16..31 (contiguous).
__device__ __forceinline__ v16bf load_b_frag(const __bf16* __restrict__ W,
                                             int n0, int k0, int ld, int lane) {
  int n    = n0 + (lane & 15);
  int koff = k0 + ((lane >> 4) << 4);
  return *(const v16bf*)(W + (size_t)n * ld + koff);
}

__global__ void cast_f32_to_bf16(const float* __restrict__ in, __bf16* __restrict__ out, int n) {
  int i = (blockIdx.x * blockDim.x + threadIdx.x) * 4;
  if (i + 3 < n) {
    float4 v = *(const float4*)(in + i);
    v4bf o;
    o[0] = (__bf16)v.x; o[1] = (__bf16)v.y; o[2] = (__bf16)v.z; o[3] = (__bf16)v.w;
    *(v4bf*)(out + i) = o;
  }
}

// Fragment set for one 64x32 wave tile at one K step.
struct Frags {
  v16bf a[4];
  v16bf b0, b1;
};

__device__ __forceinline__ void load_frags(Frags& f,
                                           const __bf16* __restrict__ A,
                                           const __bf16* __restrict__ W,
                                           int m0, int n0, int k0, int K, int lane) {
#pragma unroll
  for (int i = 0; i < 4; ++i) f.a[i] = load_a_frag(A, m0 + i * 16, k0, K, lane);
  f.b0 = load_b_frag(W, n0,      k0, K, lane);
  f.b1 = load_b_frag(W, n0 + 16, k0, K, lane);
}

__device__ __forceinline__ void mma_frags(v8f acc[4][2], const Frags& f) {
#pragma unroll
  for (int i = 0; i < 4; ++i) {
    acc[i][0] = WMMA_BF16(f.a[i], f.b0, acc[i][0]);
    acc[i][1] = WMMA_BF16(f.a[i], f.b1, acc[i][1]);
  }
}

// C = (A[MxK] . W[NxK]^T) * scale.  Exactly one of Cf / Cb / Ct is non-null:
//   Cf : fp32 row-major MxN ; Cb : bf16 row-major MxN ; Ct : bf16 TRANSPOSED (NxM).
// Block = 256 threads = 8 waves arranged 2(M) x 4(N); block tile 128x128.
// Wave tile 64(M) x 32(N), ping-pong double buffered over K (step 64):
// two statically-named fragment sets alternate, so there are no rotation
// copies and loads overlap WMMAs with partial s_wait_loadcnt.
__global__ __launch_bounds__(256) void gemm_bf16_wmma(
    const __bf16* __restrict__ A, const __bf16* __restrict__ W,
    float* __restrict__ Cf, __bf16* __restrict__ Cb, __bf16* __restrict__ Ct,
    int M, int N, int K, float scale) {
  int lane = threadIdx.x & 31;
  int wave = threadIdx.x >> 5;
  int m0 = blockIdx.x * 128 + (wave >> 2) * 64;
  int n0 = blockIdx.y * 128 + (wave & 3) * 32;

  v8f acc[4][2];
#pragma unroll
  for (int i = 0; i < 4; ++i)
#pragma unroll
    for (int j = 0; j < 2; ++j) acc[i][j] = v8f_zero();

  Frags f0, f1;
  load_frags(f0, A, W, m0, n0, 0, K, lane);

  int k0 = 0;
  for (; k0 < K - 64; k0 += 64) {
    load_frags(f1, A, W, m0, n0, k0 + 32, K, lane);
    mma_frags(acc, f0);
    load_frags(f0, A, W, m0, n0, k0 + 64, K, lane);
    mma_frags(acc, f1);
  }
  // tail: K multiple of 64 -> exactly two K-steps left (k0 and k0+32)
  load_frags(f1, A, W, m0, n0, k0 + 32, K, lane);
  mma_frags(acc, f0);
  mma_frags(acc, f1);

  int cl = lane & 15, hi = lane >> 4;
#pragma unroll
  for (int i = 0; i < 4; ++i)
#pragma unroll
    for (int j = 0; j < 2; ++j) {
      int colg = n0 + j * 16 + cl;
      if (Ct) {  // transposed store: per-lane 8 consecutive M entries -> one 16B store
        v8bf pk;
#pragma unroll
        for (int r = 0; r < 8; ++r) pk[r] = (__bf16)(acc[i][j][r] * scale);
        *(v8bf*)(Ct + (size_t)colg * M + m0 + i * 16 + hi * 8) = pk;
      } else {
#pragma unroll
        for (int r = 0; r < 8; ++r) {
          int rowg = m0 + i * 16 + r + hi * 8;
          float v = acc[i][j][r] * scale;
          if (Cf) Cf[(size_t)rowg * N + colg] = v;
          else    Cb[(size_t)rowg * N + colg] = (__bf16)v;
        }
      }
    }
}

// Flash attention. Q,K row-major [B*S][D] (head columns h*128..), V transposed [D][B*S].
// Block = 8 waves; each wave owns a 16-row Q tile, iterates causal K/V blocks of 32 tokens.
__global__ __launch_bounds__(256) void flash_attn_wmma(
    const __bf16* __restrict__ Q, const __bf16* __restrict__ Kmat,
    const __bf16* __restrict__ Vt, __bf16* __restrict__ O) {
  __shared__ __align__(32) __bf16 plds[8][16 * 32];  // 1 KB per wave: P tile bounce

  int lane = threadIdx.x & 31;
  int wave = threadIdx.x >> 5;
  int cl = lane & 15, hi = lane >> 4;
  int b = blockIdx.z, h = blockIdx.y;
  int q0 = blockIdx.x * 128 + wave * 16;  // query row (within sequence) of this wave

  const __bf16* Qh = Q    + (size_t)b * Sc * Dc + h * HDc;
  const __bf16* Kh = Kmat + (size_t)b * Sc * Dc + h * HDc;
  const __bf16* Vh = Vt   + (size_t)(h * HDc) * (Bc * Sc) + (size_t)b * Sc;  // row stride B*S

  // Q fragments for all of head-dim (4 x K=32), reused across every key block.
  v16bf qa[4];
#pragma unroll
  for (int t = 0; t < 4; ++t) qa[t] = load_a_frag(Qh, q0, t * 32, Dc, lane);

  v8f acc[8];
  float mrow[8], lrow[8];
#pragma unroll
  for (int t = 0; t < 8; ++t) acc[t] = v8f_zero();
#pragma unroll
  for (int r = 0; r < 8; ++r) { mrow[r] = -3.0e38f; lrow[r] = 0.0f; }

  int nkb = q0 / 32 + 1;  // causal: only key blocks with start <= last query row
  for (int kb = 0; kb < nkb; ++kb) {
    int t0 = kb * 32;
    // S tile 16x32 = two 16x16 WMMAs, contraction over head dim (4 x K=32)
    v8f s0 = v8f_zero(), s1 = v8f_zero();
#pragma unroll
    for (int t = 0; t < 4; ++t) {
      v16bf kb0 = load_b_frag(Kh, t0,      t * 32, Dc, lane);
      v16bf kb1 = load_b_frag(Kh, t0 + 16, t * 32, Dc, lane);
      s0 = WMMA_BF16(qa[t], kb0, s0);
      s1 = WMMA_BF16(qa[t], kb1, s1);
    }
    // mask + online softmax (row r+8*hi lives across lanes of this 16-lane half)
    float alpha[8];
#pragma unroll
    for (int r = 0; r < 8; ++r) {
      int row = q0 + r + hi * 8;
      if (t0 + cl      > row) s0[r] = -1.0e9f;
      if (t0 + 16 + cl > row) s1[r] = -1.0e9f;
      float mx = fmaxf(s0[r], s1[r]);
      mx = fmaxf(mx, __shfl_xor(mx, 1));
      mx = fmaxf(mx, __shfl_xor(mx, 2));
      mx = fmaxf(mx, __shfl_xor(mx, 4));
      mx = fmaxf(mx, __shfl_xor(mx, 8));
      float mnew = fmaxf(mrow[r], mx);
      alpha[r] = __expf(mrow[r] - mnew);
      mrow[r] = mnew;
      float p0 = __expf(s0[r] - mnew);
      float p1 = __expf(s1[r] - mnew);
      float rs = p0 + p1;
      rs += __shfl_xor(rs, 1);
      rs += __shfl_xor(rs, 2);
      rs += __shfl_xor(rs, 4);
      rs += __shfl_xor(rs, 8);
      lrow[r] = lrow[r] * alpha[r] + rs;
      s0[r] = p0; s1[r] = p1;
    }
#pragma unroll
    for (int t = 0; t < 8; ++t)
#pragma unroll
      for (int r = 0; r < 8; ++r) acc[t][r] *= alpha[r];

    // C-layout P (f32) -> row-major bf16 16x32 in LDS -> reload in A-fragment layout
    __bf16* pw = &plds[wave][0];
#pragma unroll
    for (int r = 0; r < 8; ++r) {
      int row = r + hi * 8;
      pw[row * 32 + cl]      = (__bf16)s0[r];
      pw[row * 32 + 16 + cl] = (__bf16)s1[r];
    }
    asm volatile("s_wait_dscnt 0" ::: "memory");
    v16bf pa;
    {
      const __bf16* pr = pw + (lane & 15) * 32 + ((lane >> 4) << 4);
      v8bf lo = *(const v8bf*)pr;
      v8bf hh = *(const v8bf*)(pr + 8);
#pragma unroll
      for (int i = 0; i < 8; ++i) { pa[i] = lo[i]; pa[i + 8] = hh[i]; }
    }
    // acc(16x128) += P(16x32) @ V(32x128); V^T gives contiguous B fragments
#pragma unroll
    for (int t = 0; t < 8; ++t) {
      v16bf bv = *(const v16bf*)(Vh + (size_t)(t * 16 + cl) * (Bc * Sc) + t0 + (hi << 4));
      acc[t] = WMMA_BF16(pa, bv, acc[t]);
    }
  }

#pragma unroll
  for (int r = 0; r < 8; ++r) lrow[r] = 1.0f / lrow[r];
  __bf16* Oh = O + (size_t)b * Sc * Dc + h * HDc;
#pragma unroll
  for (int t = 0; t < 8; ++t)
#pragma unroll
    for (int r = 0; r < 8; ++r)
      Oh[(size_t)(q0 + r + hi * 8) * Dc + t * 16 + cl] = (__bf16)(acc[t][r] * lrow[r]);
}

extern "C" void kernel_launch(void* const* d_in, const int* in_sizes, int n_in,
                              void* d_out, int out_size, void* d_ws, size_t ws_size,
                              hipStream_t stream) {
  const float* x  = (const float*)d_in[0];
  // d_in[1] = attn_mask: exactly the causal mask; applied analytically in-kernel.
  const float* Wq = (const float*)d_in[2];
  const float* Wk = (const float*)d_in[3];
  const float* Wv = (const float*)d_in[4];
  const float* Wo = (const float*)d_in[5];
  float* out = (float*)d_out;

  const size_t nx = (size_t)Bc * Sc * Dc;  // 8,388,608
  const size_t nw = (size_t)Dc * Dc;       // 4,194,304

  __bf16* xb  = (__bf16*)d_ws;
  __bf16* wqb = xb  + nx;
  __bf16* wkb = wqb + nw;
  __bf16* wvb = wkb + nw;
  __bf16* wob = wvb + nw;
  __bf16* Qb  = wob + nw;
  __bf16* Kb  = Qb  + nx;
  __bf16* Vt  = Kb  + nx;   // transposed: [D][B*S]
  __bf16* Yb  = Vt  + nx;   // total ws use ~117 MB

  dim3 blk(256);
  cast_f32_to_bf16<<<(int)(nx / 1024), blk, 0, stream>>>(x,  xb,  (int)nx);
  cast_f32_to_bf16<<<(int)(nw / 1024), blk, 0, stream>>>(Wq, wqb, (int)nw);
  cast_f32_to_bf16<<<(int)(nw / 1024), blk, 0, stream>>>(Wk, wkb, (int)nw);
  cast_f32_to_bf16<<<(int)(nw / 1024), blk, 0, stream>>>(Wv, wvb, (int)nw);
  cast_f32_to_bf16<<<(int)(nw / 1024), blk, 0, stream>>>(Wo, wob, (int)nw);

  const int M = Bc * Sc;  // 4096
  dim3 gp(M / 128, Dc / 128);  // 32 x 16, 8 waves each
  const float qscale = 0.088388347648318447f;  // 1/sqrt(128), folded into Q projection
  gemm_bf16_wmma<<<gp, blk, 0, stream>>>(xb, wqb, nullptr, Qb, nullptr, M, Dc, Dc, qscale);
  gemm_bf16_wmma<<<gp, blk, 0, stream>>>(xb, wkb, nullptr, Kb, nullptr, M, Dc, Dc, 1.0f);
  gemm_bf16_wmma<<<gp, blk, 0, stream>>>(xb, wvb, nullptr, nullptr, Vt, M, Dc, Dc, 1.0f);

  flash_attn_wmma<<<dim3(Sc / 128, Hc, Bc), blk, 0, stream>>>(Qb, Kb, Vt, Yb);

  gemm_bf16_wmma<<<gp, blk, 0, stream>>>(Yb, wob, out, nullptr, nullptr, M, Dc, Dc, 1.0f);
}
